// DGPLoss_56203942035876
// MI455X (gfx1250) — matrix operand order repaired and checked by reading, not compiled, not run on hardware
//
#include <hip/hip_runtime.h>
#include <hip/hip_bf16.h>

typedef __attribute__((ext_vector_type(16))) _Float16 v16h;
typedef __attribute__((ext_vector_type(8)))  float    v8f;

#define BATCH 2
#define CHAN  32
#define HH    192
#define WW    192
#define HP    188              // valid centers per dim (k=5 -> 192-4)
#define SEGS  16               // ceil(188/12) column segments of 12 centers
#define ITEMS (BATCH * HP * SEGS)          // 6016 wave work-items
#define WPB   8                             // waves per 256-thread block
#define MAIN_BLOCKS (ITEMS / WPB)           // 752
#define EPSV  1e-8f
#define INV_TAU 0.1f

// workspace layout (bytes)
#define OFS_F16   0u                                   // 2*192*192*32 halves = 4,718,592 B
#define OFS_NSQ   4718592u                             // 2*192*192 f32      =   294,912 B
#define OFS_INVD  5013504u                             // 2*192*192 f32      =   294,912 B
#define OFS_PART  5308416u                             // 752 f32
// total ~5.31 MB

// ---------------------------------------------------------------------------
// Kernel 1: pack features pixel-major f16 (c contiguous), per-pixel squared
// norm (computed from the f16-rounded values so sd2 = nA+nB-2*dot >= 0 up to
// accumulator rounding), and inverse depth 1/(d+1e-6).
// ---------------------------------------------------------------------------
__global__ __launch_bounds__(256) void dgp_prep(const float* __restrict__ feat,
                                                const float* __restrict__ depth,
                                                _Float16* __restrict__ f16,
                                                float* __restrict__ nsq,
                                                float* __restrict__ invd) {
    int p = blockIdx.x * blockDim.x + threadIdx.x;
    if (p >= BATCH * HH * WW) return;
    int b  = p / (HH * WW);
    int hw = p - b * (HH * WW);
    const float* src = feat + (size_t)b * CHAN * HH * WW + hw;
    _Float16* dst = f16 + (size_t)p * CHAN;
    float s = 0.f;
#pragma unroll
    for (int c = 0; c < CHAN; ++c) {
        float v = src[(size_t)c * HH * WW];
        _Float16 h = (_Float16)v;
        dst[c] = h;
        float vr = (float)h;
        s = fmaf(vr, vr, s);
    }
    nsq[p]  = s;
    invd[p] = 1.0f / (depth[p] + 1e-6f);
}

// ---------------------------------------------------------------------------
// Kernel 2: one wave per (batch, row, 12-center column segment).
// A = f16 feats of pixels (r, c0..c0+15)          [16 x 32]
// B = f16 feats of pixels (r+dr, c0-2..c0+13)     [32 x 16]
// D = A*B : D[m,n] = dot(center m, neighbor at dc = n-m-2). Diagonals
// k = n-m in [0,4] are harvested for 12 centers (m<=11); 5 WMMAs cover dr.
// ---------------------------------------------------------------------------
__global__ __launch_bounds__(256) void dgp_main(const _Float16* __restrict__ f16,
                                                const float* __restrict__ nsq,
                                                const float* __restrict__ invd,
                                                float* __restrict__ partials) {
    const int lane = threadIdx.x & 31;
    const int wave = threadIdx.x >> 5;
    const int item = blockIdx.x * WPB + wave;

    const int b   = item / (HP * SEGS);
    const int rem = item - b * (HP * SEGS);
    const int r   = 2 + rem / SEGS;
    const int c0  = 2 + (rem % SEGS) * 12;

    const int nn = lane & 15;          // column index N of D; also A row M id
    const int hi = lane >> 4;          // lane half selects K sub-range

    // ---- A fragment: lane holds row M = nn (pixel col c0+nn, clamped).
    // A layout: halves 0..15 = K 16*hi..16*hi+15 (matches B). Pixel storage is
    // channel-contiguous, so this is one 32-byte contiguous chunk.
    {
    }
    const int acol = min(c0 + nn, WW - 1);
    const _Float16* ap = f16 + (((size_t)(b * HH + r)) * WW + acol) * CHAN + hi * 16;
    v16h a;
#pragma unroll
    for (int i = 0; i < 8; ++i) { a[i] = ap[i]; a[8 + i] = ap[8 + i]; }

    // ---- per-M center data (dr-independent)
    float nsC[8], idC[8];
    int   ccl[8];
    bool  vm[8];
#pragma unroll
    for (int v = 0; v < 8; ++v) {
        int m  = v + hi * 8;
        int cc = c0 + m;
        vm[v]  = (m <= 11) && (cc <= 189);
        int ci = min(cc, WW - 1);
        ccl[v] = ci;
        nsC[v] = nsq[(b * HH + r) * WW + ci];
        idC[v] = invd[(b * HH + r) * WW + ci];
    }

    float acc = 0.f;

#pragma unroll
    for (int dr = -2; dr <= 2; ++dr) {
        const int rb = r + dr;                              // always in [0,191]
        // ---- B fragment: lane half hi holds K 16*hi..16*hi+15 of column nn
        // (pixel col c0-2+nn, clamped); channel-contiguous -> 32B chunk.
        const int qcol = min(max(c0 - 2 + nn, 0), WW - 1);
        const _Float16* bp = f16 + (((size_t)(b * HH + rb)) * WW + qcol) * CHAN + hi * 16;
        v16h bf;
#pragma unroll
        for (int i = 0; i < 16; ++i) bf[i] = bp[i];

        v8f cz = {0.f, 0.f, 0.f, 0.f, 0.f, 0.f, 0.f, 0.f};
        v8f d = __builtin_amdgcn_wmma_f32_16x16x32_f16(
            /*neg_a=*/false, a, /*neg_b=*/false, bf,
            /*c_mod=*/(short)0, cz, /*reuse_a=*/false, /*reuse_b=*/false);

        const int rowq = (b * HH + rb) * WW;
        const int posq = rb * WW;                           // batch-0 invd row
#pragma unroll
        for (int v = 0; v < 8; ++v) {
            int  m = v + hi * 8;
            int  k = nn - m;                                // dc = k-2
            bool valid = vm[v] && (k >= 0) && (k <= 4) && !((dr == 0) && (k == 2));
            int  qc = min(max(ccl[v] + (k - 2), 0), WW - 1);

            float dot = d[v];
            float nsQ = nsq[rowq + qc];
            float idQ = invd[rowq + qc];

            float sd2 = fmaxf(nsC[v] + nsQ - 2.f * dot, 0.f);
            float sd  = __builtin_sqrtf(sd2);
            float dd  = fabsf(idC[v] - idQ);

            bool  keep = (sd > EPSV) && (dd > EPSV);        // m-mask before clamp
            float sdc = fmaxf(sd, EPSV);
            float ddc = fmaxf(dd, EPSV);

            // cross-batch pos count: sum_a [invd[a, neighbor] > 0]
            float pos = (invd[posq + qc]           > 0.f ? 1.f : 0.f)
                      + (invd[HH * WW + posq + qc] > 0.f ? 1.f : 0.f);

            float l = __expf(-(ddc * INV_TAU + sdc * sdc)) * pos;
            acc += (valid && keep) ? l : 0.f;
        }
    }

    // ---- deterministic block reduction -> one partial per block
    __shared__ float red[256];
    red[threadIdx.x] = acc;
    __syncthreads();
#pragma unroll
    for (int s = 128; s > 0; s >>= 1) {
        if (threadIdx.x < s) red[threadIdx.x] += red[threadIdx.x + s];
        __syncthreads();
    }
    if (threadIdx.x == 0) partials[blockIdx.x] = red[0];
}

// ---------------------------------------------------------------------------
// Kernel 3: fixed-order final sum (deterministic across graph replays).
// Mean denominator: inner mean over 5x5 (=25), outer mean over (2,2,35344).
// ---------------------------------------------------------------------------
__global__ __launch_bounds__(256) void dgp_final(const float* __restrict__ partials,
                                                 float* __restrict__ out) {
    __shared__ float red[256];
    float s = 0.f;
    for (int i = threadIdx.x; i < MAIN_BLOCKS; i += 256) s += partials[i];
    red[threadIdx.x] = s;
    __syncthreads();
#pragma unroll
    for (int t = 128; t > 0; t >>= 1) {
        if (threadIdx.x < t) red[threadIdx.x] += red[threadIdx.x + t];
        __syncthreads();
    }
    if (threadIdx.x == 0)
        out[0] = red[0] * (1.0f / (25.0f * 4.0f * (float)(HP * HP)));
}

extern "C" void kernel_launch(void* const* d_in, const int* in_sizes, int n_in,
                              void* d_out, int out_size, void* d_ws, size_t ws_size,
                              hipStream_t stream) {
    const float* sem = (const float*)d_in[0];   // (2,32,192,192) f32
    const float* dep = (const float*)d_in[1];   // (2,1,192,192)  f32

    char* ws = (char*)d_ws;                     // needs ~5.31 MB
    _Float16* f16   = (_Float16*)(ws + OFS_F16);
    float*    nsq   = (float*)(ws + OFS_NSQ);
    float*    invd  = (float*)(ws + OFS_INVD);
    float*    parts = (float*)(ws + OFS_PART);

    const int npix = BATCH * HH * WW;           // 73728
    dgp_prep<<<(npix + 255) / 256, 256, 0, stream>>>(sem, dep, f16, nsq, invd);
    dgp_main<<<MAIN_BLOCKS, 256, 0, stream>>>(f16, nsq, invd, parts);
    dgp_final<<<1, 256, 0, stream>>>(parts, (float*)d_out);
}
